// EnergyPredictor_11433202942436
// MI455X (gfx1250) — matrix operand order repaired and checked by reading, not compiled, not run on hardware
//
#include <hip/hip_runtime.h>
#include <hip/hip_bf16.h>
#include <stdint.h>

// ---------------------------------------------------------------------------
// CDNA5 / gfx1250: wave32, WMMA 16x16x32 bf16 (f32 accumulate)
// ---------------------------------------------------------------------------
typedef __attribute__((ext_vector_type(16))) __bf16 v16bf;
typedef __attribute__((ext_vector_type(8)))  float  v8f;

#define DEV static __device__ __forceinline__

DEV int imin(int a, int b) { return a < b ? a : b; }

// Native f32 -> bf16 (lowers to v_cvt_pk_bf16_f32 on gfx1250)
DEV uint32_t packbf(float lo, float hi) {
  union { uint32_t u; __bf16 h[2]; } r;
  r.h[0] = (__bf16)lo;
  r.h[1] = (__bf16)hi;
  return r.u;
}

// Packed bf16 multiply via CDNA5 V_PK_MUL_BF16 (VOP3P) — no f32 round trip.
DEV uint32_t pkmul_bf(uint32_t a, uint32_t b) {
  uint32_t d;
  asm("v_pk_mul_bf16 %0, %1, %2" : "=v"(d) : "v"(a), "v"(b));
  return d;
}

DEV float silu(float x) { return x / (1.0f + __expf(-x)); }

union AFrag { v16bf v; uint32_t u[8]; __bf16 h[16]; };

// K-offset of dword pair r for this half-lane within a 16-bit 16x32 A fragment
// (ISA 7.12.2: lanes 0-15 hold K 0..7,16..23; lanes 16-31 hold K 8..15,24..31)
DEV int a_koff(int r, int hi) { return ((r & 3) << 1) + (hi << 3) + ((r >> 2) << 4); }

// B fragment pre-swizzled in memory: frag = 32 lanes x 16 bf16, lane-contiguous.
DEV v16bf load_bfrag(const __bf16* w, int frag, int lane) {
  return *(const v16bf*)(w + (size_t)frag * 512 + lane * 16);
}

// A fragment from a bf16 row in global memory: two contiguous 16B runs per lane.
DEV v16bf load_afrag_row(const uint32_t* rowp, int ks, int hi) {
  union { v16bf v; uint4 q[2]; } u;
  const uint32_t* p = rowp + (ks << 4) + (hi << 2);
  u.q[0] = *(const uint4*)(p);       // K = ks*32 + 8*hi + [0..7]
  u.q[1] = *(const uint4*)(p + 8);   // K = ks*32 + 8*hi + 16 + [0..7]
  return u.v;
}

// A fragment from a bf16 LDS tile (row-major, ld=64): two ds_load_b128, no VALU.
DEV v16bf load_afrag_ldsbf(const __bf16* buf, int nl, int hi, int kbase) {
  union { v16bf v; uint4 q[2]; } u;
  const uint32_t* p = (const uint32_t*)buf + nl * 32 + (kbase >> 1) + (hi << 2);
  u.q[0] = *(const uint4*)(p);
  u.q[1] = *(const uint4*)(p + 8);
  return u.v;
}

// Same, with per-row bf16 scale (s broadcast in both halves of s2).
DEV v16bf load_afrag_ldsbf_scaled(const __bf16* buf, int nl, int hi, int kbase,
                                  uint32_t s2) {
  AFrag f; f.v = load_afrag_ldsbf(buf, nl, hi, kbase);
#pragma unroll
  for (int r = 0; r < 8; ++r) f.u[r] = pkmul_bf(f.u[r], s2);
  return f.v;
}

DEV v8f wmma_bf16(v16bf a, v16bf b, v8f c) {
  return __builtin_amdgcn_wmma_f32_16x16x32_bf16(false, a, false, b, (short)0, c,
                                                 false, false);
}

#define LDS_FENCE() asm volatile("s_wait_dscnt 0" ::: "memory")

// ---------------------------------------------------------------------------
// Weight prep: f32 -> bf16, swizzled to B-fragment layout.
// Logical element (k,n) of a [K x Nn] slice at src[k*rowStride + n].
// B frag layout: lane = n + 16*((k%16)>=8), element idx = (k%8) + 8*((k%32)>=16).
// ---------------------------------------------------------------------------
__global__ void swizzle_b(const float* __restrict__ src, __bf16* __restrict__ dst,
                          int K, int Nn, int rowStride, int nNt, int total) {
  int t = blockIdx.x * blockDim.x + threadIdx.x;
  if (t >= total) return;
  int frag = t >> 9, j = t & 511;
  int lane = j >> 4, idx = j & 15;
  int r = idx >> 1, half = idx & 1;
  int kt = frag / nNt, nt = frag - kt * nNt;
  int n = (nt << 4) + (lane & 15);
  int k = (kt << 5) + ((lane >> 4) << 3) + (idx & 7) + ((idx >> 3) << 4);
  float v = (k < K && n < Nn) ? src[(size_t)k * rowStride + n] : 0.0f;
  dst[t] = (__bf16)v;
}

// f32 features -> zero-padded bf16 rows (optionally through silu gate)
__global__ void convert_h_kernel(const float* __restrict__ hin, int ldin, int di,
                                 __bf16* __restrict__ hb, int Kd, int Nn,
                                 int applySilu) {
  int t = blockIdx.x * blockDim.x + threadIdx.x;
  if (t >= Nn * Kd) return;
  int row = t / Kd, k = t - row * Kd;
  float v = 0.0f;
  if (k < di) {
    v = hin[(size_t)row * ldin + k];
    if (applySilu) v = silu(v);
  }
  hb[t] = (__bf16)v;
}

// ---------------------------------------------------------------------------
// Self-connection: sc[n,o] = sum_a attr[n,a] * (h[n,:] @ W_sc[:,a,:])[o]
// One wave per (16-node tile, 16-col tile). Writes (not adds) into hout.
// ---------------------------------------------------------------------------
__global__ void sc_kernel(const __bf16* __restrict__ hb,
                          const float* __restrict__ attr,
                          const __bf16* __restrict__ wsc,
                          float* __restrict__ hout,
                          int Nn, int Kd, int dOut, int nNt) {
  int wid  = blockIdx.x * (blockDim.x >> 5) + (threadIdx.x >> 5);
  int lane = threadIdx.x & 31, nl = lane & 15, hi = lane >> 4;
  int tiles = (Nn >> 4) * nNt;
  if (wid >= tiles) return;
  int tb = wid / nNt, nt = wid - tb * nNt;
  int m0 = tb << 4;

  uint32_t av2[8];                       // attr scale, bf16-broadcast per channel
  const float* ap = attr + (size_t)(m0 + nl) * 8;
#pragma unroll
  for (int a = 0; a < 8; ++a) { float s = ap[a]; av2[a] = packbf(s, s); }

  const uint32_t* rowp = (const uint32_t*)(hb + (size_t)(m0 + nl) * Kd);
  int nK = Kd >> 5;
  v8f c = {};
  for (int ks = 0; ks < nK; ++ks) {
    AFrag raw; raw.v = load_afrag_row(rowp, ks, hi);
#pragma unroll
    for (int a = 0; a < 8; ++a) {
      AFrag fa;
#pragma unroll
      for (int r = 0; r < 8; ++r) fa.u[r] = pkmul_bf(raw.u[r], av2[a]);
      c = wmma_bf16(fa.v, load_bfrag(wsc, (a * nK + ks) * nNt + nt, lane), c);
    }
  }
  int col = (nt << 4) + nl;
  if (col < dOut) {
#pragma unroll
    for (int g = 0; g < 8; ++g)
      hout[(size_t)(m0 + g + (hi << 3)) * 144 + col] = c[g];
  }
}

// ---------------------------------------------------------------------------
// Fused per-edge pipeline: radial MLP -> gathered feat GEMM -> radial
// weighting -> tensor product -> atomic scatter-add aggregation.
// One wave per 16-edge tile; per-wave bf16 LDS tiles for the C->A
// layout transposes between WMMA stages.
// ---------------------------------------------------------------------------
struct alignas(16) WaveLds {
  __bf16 x[16 * 64];   // h1 / feat tiles (bf16)
  __bf16 y[16 * 64];   // h2 tile (bf16)
  float  w[16 * 64];   // radial weights (f32)
};

template <int NNT>
__global__ __launch_bounds__(128)
void edge_kernel(const float* __restrict__ emb,
                 const __bf16* __restrict__ wfc1,
                 const __bf16* __restrict__ wfc2,
                 const __bf16* __restrict__ wfc3,
                 const __bf16* __restrict__ hb,
                 const __bf16* __restrict__ wlin,
                 const float* __restrict__ eattr,
                 const __bf16* __restrict__ wtp,
                 const int* __restrict__ esrc,
                 const int* __restrict__ edst,
                 float* __restrict__ hout,
                 int E, int Kd, int dOut) {
  __shared__ WaveLds smem[4];
  int wave = threadIdx.x >> 5;
  int lane = threadIdx.x & 31, nl = lane & 15, hi = lane >> 4;
  int wid = blockIdx.x * 4 + wave;
  int tiles = (E + 15) >> 4;
  if (wid >= tiles) return;
  int e0 = wid << 4;
  int erow = imin(e0 + nl, E - 1);
  __bf16* bufX = smem[wave].x;
  __bf16* bufY = smem[wave].y;
  float*  wbuf = smem[wave].w;

  __builtin_prefetch(wtp, 0, 3);   // global_prefetch: warm tensor-product weights

  // ---- radial MLP stage 1: h1 = silu(emb @ W_fc1), K=10 padded to 32 ----
  {
    const float* ep = emb + (size_t)erow * 10;
    AFrag a1;
#pragma unroll
    for (int r = 0; r < 8; ++r) {
      int k = a_koff(r, hi);
      float x0 = (k < 10)     ? ep[k]     : 0.0f;
      float x1 = (k + 1 < 10) ? ep[k + 1] : 0.0f;
      a1.u[r] = packbf(x0, x1);
    }
#pragma unroll
    for (int nt = 0; nt < 2; ++nt) {
      v8f c = {};
      c = wmma_bf16(a1.v, load_bfrag(wfc1, nt, lane), c);
#pragma unroll
      for (int g = 0; g < 8; ++g)
        bufX[(g + (hi << 3)) * 64 + (nt << 4) + nl] = (__bf16)silu(c[g]);
    }
  }
  LDS_FENCE();
  // ---- stage 2: h2 = silu(h1 @ W_fc2), K=32 ----
  {
    v16bf a2 = load_afrag_ldsbf(bufX, nl, hi, 0);
#pragma unroll
    for (int nt = 0; nt < 4; ++nt) {
      v8f c = {};
      c = wmma_bf16(a2, load_bfrag(wfc2, nt, lane), c);
#pragma unroll
      for (int g = 0; g < 8; ++g)
        bufY[(g + (hi << 3)) * 64 + (nt << 4) + nl] = (__bf16)silu(c[g]);
    }
  }
  LDS_FENCE();
  // ---- stage 3: w = h2 @ W_fc3, K=64 (2 k-steps); w -> wbuf (f32) ----
  {
    v16bf a30 = load_afrag_ldsbf(bufY, nl, hi, 0);
    v16bf a31 = load_afrag_ldsbf(bufY, nl, hi, 32);
#pragma unroll
    for (int nt = 0; nt < 4; ++nt) {
      v8f c = {};
      c = wmma_bf16(a30, load_bfrag(wfc3, nt,     lane), c);
      c = wmma_bf16(a31, load_bfrag(wfc3, 4 + nt, lane), c);
#pragma unroll
      for (int g = 0; g < 8; ++g)
        wbuf[(g + (hi << 3)) * 64 + (nt << 4) + nl] = c[g];
    }
  }
  // ---- feat = (h[src] @ W_lin) * w ; store bf16 to bufX ----
  {
    int src = esrc[erow];
    const uint32_t* rowp = (const uint32_t*)(hb + (size_t)src * Kd);
    int nK = Kd >> 5;
    v8f cf0 = {}, cf1 = {}, cf2 = {}, cf3 = {};
    for (int ks = 0; ks < nK; ++ks) {
      v16bf fa = load_afrag_row(rowp, ks, hi);
      cf0 = wmma_bf16(fa, load_bfrag(wlin, ks * 4 + 0, lane), cf0);
      cf1 = wmma_bf16(fa, load_bfrag(wlin, ks * 4 + 1, lane), cf1);
      cf2 = wmma_bf16(fa, load_bfrag(wlin, ks * 4 + 2, lane), cf2);
      cf3 = wmma_bf16(fa, load_bfrag(wlin, ks * 4 + 3, lane), cf3);
    }
    LDS_FENCE();
#pragma unroll
    for (int g = 0; g < 8; ++g) {
      int m = g + (hi << 3);
      bufX[m * 64 +  0 + nl] = (__bf16)(cf0[g] * wbuf[m * 64 +  0 + nl]);
      bufX[m * 64 + 16 + nl] = (__bf16)(cf1[g] * wbuf[m * 64 + 16 + nl]);
      bufX[m * 64 + 32 + nl] = (__bf16)(cf2[g] * wbuf[m * 64 + 32 + nl]);
      bufX[m * 64 + 48 + nl] = (__bf16)(cf3[g] * wbuf[m * 64 + 48 + nl]);
    }
  }
  LDS_FENCE();
  // ---- msg = sum_a diag(attr[:,a]) . feat @ W_tp[:,a,:]; atomic aggregate ----
  {
    uint32_t av2[9];
    const float* ap = eattr + (size_t)erow * 9;
#pragma unroll
    for (int a = 0; a < 9; ++a) { float s = ap[a]; av2[a] = packbf(s, s); }
    int dstm[8];
#pragma unroll
    for (int g = 0; g < 8; ++g)
      dstm[g] = edst[imin(e0 + g + (hi << 3), E - 1)];

    v8f acc[NNT] = {};
#pragma unroll
    for (int a = 0; a < 9; ++a) {
      v16bf f0 = load_afrag_ldsbf_scaled(bufX, nl, hi, 0,  av2[a]);
      v16bf f1 = load_afrag_ldsbf_scaled(bufX, nl, hi, 32, av2[a]);
#pragma unroll
      for (int nt = 0; nt < NNT; ++nt) {
        acc[nt] = wmma_bf16(f0, load_bfrag(wtp, (a * 2 + 0) * NNT + nt, lane), acc[nt]);
        acc[nt] = wmma_bf16(f1, load_bfrag(wtp, (a * 2 + 1) * NNT + nt, lane), acc[nt]);
      }
    }
    const float invDeg = 0.25f;   // 1/sqrt(NUM_NEIGHBORS)
#pragma unroll
    for (int nt = 0; nt < NNT; ++nt) {
      int col = (nt << 4) + nl;
      if (col < dOut) {
#pragma unroll
        for (int g = 0; g < 8; ++g)
          if (e0 + g + (hi << 3) < E)
            atomicAdd(&hout[(size_t)dstm[g] * 144 + col], acc[nt][g] * invDeg);
      }
    }
  }
}

// ---------------------------------------------------------------------------
// Readout + softmax
// ---------------------------------------------------------------------------
__global__ void zero_kernel(float* p, int n) {
  int t = blockIdx.x * blockDim.x + threadIdx.x;
  if (t < n) p[t] = 0.0f;
}
__global__ void readout_kernel(const float* __restrict__ h, const int* __restrict__ batch,
                               float* __restrict__ g, int Nn) {
  int n = blockIdx.x * blockDim.x + threadIdx.x;
  if (n >= Nn) return;
  int b = batch[n];
  const float* hp = h + (size_t)n * 144;
#pragma unroll
  for (int o = 0; o < 10; ++o) atomicAdd(&g[b * 10 + o], hp[o]);
}
__global__ void softmax_kernel(const float* __restrict__ g, float* __restrict__ out) {
  int b = threadIdx.x;
  if (b >= 64) return;
  float v[10], mx = -1e30f;
#pragma unroll
  for (int o = 0; o < 10; ++o) { v[o] = g[b * 10 + o]; mx = fmaxf(mx, v[o]); }
  float sum = 0.0f;
#pragma unroll
  for (int o = 0; o < 10; ++o) { v[o] = __expf(v[o] - mx); sum += v[o]; }
  float inv = 1.0f / sum;
#pragma unroll
  for (int o = 0; o < 10; ++o) out[b * 10 + o] = v[o] * inv;
}

// ---------------------------------------------------------------------------
// Host driver
// ---------------------------------------------------------------------------
extern "C" void kernel_launch(void* const* d_in, const int* in_sizes, int n_in,
                              void* d_out, int out_size, void* d_ws, size_t ws_size,
                              hipStream_t stream) {
  const int N = in_sizes[0] / 16;
  const int E = in_sizes[2];
  const float* x     = (const float*)d_in[0];
  const float* attr  = (const float*)d_in[1];
  const int*   esrc  = (const int*)d_in[2];
  const int*   edst  = (const int*)d_in[3];
  const float* eattr = (const float*)d_in[4];
  const float* emb   = (const float*)d_in[5];
  const int*   batch = (const int*)d_in[6];

  static const int DI[4] = {16, 144, 144, 144};
  static const int DO[4] = {144, 144, 144, 10};

  // Identify params inside each layer's group of 6 by element count
  // (robust to dict-key flattening order).
  const float *Wsc[4], *Wfc1[4], *Wfc2[4], *Wfc3[4], *Wlin[4], *Wtp[4];
  for (int l = 0; l < 4; ++l) {
    for (int j = 0; j < 6; ++j) {
      int idx = 7 + l * 6 + j;
      int sz = in_sizes[idx];
      const float* p = (const float*)d_in[idx];
      if      (sz == DI[l] * 8 * DO[l]) Wsc[l]  = p;
      else if (sz == 10 * 32)           Wfc1[l] = p;
      else if (sz == 32 * 64)           Wfc2[l] = p;
      else if (sz == 64 * 64)           Wfc3[l] = p;
      else if (sz == DI[l] * 64)        Wlin[l] = p;
      else if (sz == 64 * 9 * DO[l])    Wtp[l]  = p;
    }
  }

  char* wsp = (char*)d_ws;
  size_t off = 0;
  auto carve = [&](size_t bytes) -> void* {
    void* p = wsp + off;
    off = (off + bytes + 255) & ~(size_t)255;
    return p;
  };
  float*  hf   = (float*)carve((size_t)N * 144 * sizeof(float));
  __bf16* hbuf = (__bf16*)carve((size_t)N * 160 * sizeof(__bf16));
  float*  gbuf = (float*)carve(64 * 10 * sizeof(float));

  auto swz = [&](const float* src, int K, int Nn, int rowStride, int nK, int nNt)
      -> __bf16* {
    __bf16* dst = (__bf16*)(wsp + off);
    int total = nK * nNt * 512;                 // bf16 elems; multiple of 512 -> 1KB
    off += (size_t)total * sizeof(__bf16);
    swizzle_b<<<(total + 255) / 256, 256, 0, stream>>>(src, dst, K, Nn, rowStride, nNt, total);
    return dst;
  };

  __bf16 *wscS[4], *wfc1S[4], *wfc2S[4], *wfc3S[4], *wlinS[4], *wtpS[4];
  for (int l = 0; l < 4; ++l) {
    int Kd  = (DI[l] + 31) & ~31;   // 32 or 160
    int nK  = Kd >> 5;
    int nNt = (DO[l] + 15) >> 4;
    for (int a = 0; a < 8; ++a) {   // W_sc: 8 contiguous attr-channel slices
      __bf16* p = swz(Wsc[l] + a * DO[l], DI[l], DO[l], 8 * DO[l], nK, nNt);
      if (a == 0) wscS[l] = p;
    }
    wfc1S[l] = swz(Wfc1[l], 10, 32, 32, 1, 2);
    wfc2S[l] = swz(Wfc2[l], 32, 64, 64, 1, 4);
    wfc3S[l] = swz(Wfc3[l], 64, 64, 64, 2, 4);
    wlinS[l] = swz(Wlin[l], DI[l], 64, 64, nK, 4);
    for (int a = 0; a < 9; ++a) {   // W_tp: 9 contiguous edge-attr slices
      __bf16* p = swz(Wtp[l] + a * DO[l], 64, DO[l], 9 * DO[l], 2, nNt);
      if (a == 0) wtpS[l] = p;
    }
  }

  for (int l = 0; l < 4; ++l) {
    int Kd  = (DI[l] + 31) & ~31;
    int nNt = (DO[l] + 15) >> 4;

    {   // bf16 (silu-gated) input features, zero padded to Kd
      const float* hin = (l == 0) ? x : hf;
      int ldin = (l == 0) ? 16 : 144;
      int total = N * Kd;
      convert_h_kernel<<<(total + 255) / 256, 256, 0, stream>>>(
          hin, ldin, DI[l], hbuf, Kd, N, l > 0 ? 1 : 0);
    }
    {   // self-connection writes h_next
      int waves = (N >> 4) * nNt;
      sc_kernel<<<(waves + 7) / 8, 256, 0, stream>>>(
          hbuf, attr, wscS[l], hf, N, Kd, DO[l], nNt);
    }
    {   // fused edge pipeline adds aggregated messages
      int tiles = (E + 15) >> 4;
      int blocks = (tiles + 3) / 4;
      if (nNt == 9)
        edge_kernel<9><<<blocks, 128, 0, stream>>>(
            emb, wfc1S[l], wfc2S[l], wfc3S[l], hbuf, wlinS[l],
            eattr, wtpS[l], esrc, edst, hf, E, Kd, DO[l]);
      else
        edge_kernel<1><<<blocks, 128, 0, stream>>>(
            emb, wfc1S[l], wfc2S[l], wfc3S[l], hbuf, wlinS[l],
            eattr, wtpS[l], esrc, edst, hf, E, Kd, DO[l]);
    }
  }

  zero_kernel<<<3, 256, 0, stream>>>(gbuf, 640);
  readout_kernel<<<(N + 255) / 256, 256, 0, stream>>>(hf, batch, gbuf, N);
  softmax_kernel<<<1, 64, 0, stream>>>(gbuf, (float*)d_out);
}
